// ACTRNN_86045374808221
// MI455X (gfx1250) — matrix operand order repaired and checked by reading, not compiled, not run on hardware
//
#include <hip/hip_runtime.h>

typedef __attribute__((ext_vector_type(16))) __bf16 v16bf;
typedef __attribute__((ext_vector_type(8)))  __bf16 v8bf;
typedef __attribute__((ext_vector_type(8)))  float  v8f;

#define SEQ      512
#define BATCH    256
#define INDIM    256
#define HID      256
#define BT       16     // batch rows per workgroup (WMMA M)
#define NTHREADS 512    // 16 wave32 waves: wave w owns feature column tile j in [16w,16w+16)
#define PITCH    264    // padded bf16 row length; 528B rows -> conflict-free fragment reads

__device__ __forceinline__ v8f wmma_bf16(v16bf a, v16bf b, v8f c) {
  return __builtin_amdgcn_wmma_f32_16x16x32_bf16(false, a, false, b, (short)0, c, false, false);
}

// A fragment 16x32 bf16 from LDS tile [16][PITCH].
__device__ __forceinline__ v16bf load_a_lds(const __bf16* base, int kf, int lane) {
  const __bf16* p = base + (lane & 15) * PITCH + kf * 32 + (lane >> 4) * 16;
  v8bf lo = *(const v8bf*)(p);
  v8bf hi = *(const v8bf*)(p + 8);
  v16bf r;
#pragma unroll
  for (int i = 0; i < 8; ++i) { r[i] = lo[i]; r[i + 8] = hi[i]; }
  return r;
}

// B fragment 32x16: element i -> K = kf*32 + (lane>>4)*16 + i; row_base = &W[lane's out-feature][0]
__device__ __forceinline__ v16bf load_b_bf16(const __bf16* row_base, int kf, int lane) {
  const __bf16* p = row_base + kf * 32 + (lane >> 4) * 16;
  v8bf lo = *(const v8bf*)(p);
  v8bf hi = *(const v8bf*)(p + 8);
  v16bf r;
#pragma unroll
  for (int i = 0; i < 8; ++i) { r[i] = lo[i]; r[i + 8] = hi[i]; }
  return r;
}

__device__ __forceinline__ v16bf load_b_f32(const float* row_base, int kf, int lane) {
  const float* p = row_base + kf * 32 + (lane >> 4) * 16;
  v16bf r;
#pragma unroll
  for (int i = 0; i < 16; ++i) r[i] = (__bf16)p[i];
  return r;
}

__device__ __forceinline__ float sigmoidf_(float v) { return 1.0f / (1.0f + expf(-v)); }

// ---- init kernel: W_ih f32 [768][257] -> bf16 [768][256] in workspace ----
__global__ void wih_to_bf16(const float* __restrict__ Wih, __bf16* __restrict__ w16) {
  const int i = blockIdx.x * blockDim.x + threadIdx.x;
  if (i < 768 * 256) {
    const int r = i >> 8, c = i & 255;
    w16[i] = (__bf16)Wih[(size_t)r * 257 + c];
  }
}

__global__ __launch_bounds__(NTHREADS, 1)
void act_gru_persistent(const float* __restrict__ x,
                        const float* __restrict__ Wih,   // [768][257]
                        const float* __restrict__ Whh,   // [768][256]
                        const float* __restrict__ bih,   // [768]
                        const float* __restrict__ bhh,   // [768]
                        const float* __restrict__ Wg1,   // [256][256]
                        const float* __restrict__ bg1,   // [256]
                        const float* __restrict__ Wg2,   // [1][256]
                        const float* __restrict__ bg2,   // [1]
                        const __bf16* __restrict__ wih16,// [768][256] bf16 (workspace)
                        float* __restrict__ out)         // [512][256][256]
{
  // ---- LDS: resident weights (bf16) + tiles + lane-private gi stash ----
  __shared__ __align__(16) __bf16    s_wg1[HID * PITCH];    // Wg1 rows, 135 KB
  __shared__ __align__(16) __bf16    s_whn[HID * PITCH];    // W_hh n-gate rows, 135 KB
  __shared__ __align__(16) __bf16    s_x16[BT * PITCH];     // x_t tile (A operand)
  __shared__ __align__(16) __bf16    s_h16[BT * PITCH];     // frozen h (A operand)
  __shared__ __align__(16) __bf16    s_g16[BT * HID];       // gelu output (scalar access)
  __shared__ __align__(4)  _Float16  s_gi[NTHREADS * 24];   // lane-private gi stash (r,z,n x 8)
  __shared__ float s_w2[HID];
  __shared__ float s_halt[BT];
  __shared__ float s_pm[BT];

  const int tid  = threadIdx.x;
  const int wave = tid >> 5;
  const int lane = tid & 31;
  const int bbase = blockIdx.x * BT;

  const int jcol = wave * 16 + (lane & 15);  // this lane's feature column j in [0,256)
  const int mrow = (lane >> 4) * 8;          // C-frag: element e -> batch row mrow+e
  _Float16* gstash = s_gi + tid * 24;

  // ---- register-resident W_hh fragments for r and z gates (16 x v16bf = 128 VGPRs) ----
  v16bf whr[8], whz[8];
#pragma unroll
  for (int k = 0; k < 8; ++k) {
    whr[k] = load_b_f32(Whh + (size_t)(      jcol) * 256, k, lane);
    whz[k] = load_b_f32(Whh + (size_t)(256 + jcol) * 256, k, lane);
  }

  // per-lane scalar constants
  const float b_r  = bih[jcol]       + bhh[jcol];
  const float b_z  = bih[256 + jcol] + bhh[256 + jcol];
  const float bi_n = bih[512 + jcol];
  const float bh_n = bhh[512 + jcol];
  const float wf_r = Wih[(size_t)(      jcol) * 257 + 256];
  const float wf_z = Wih[(size_t)(256 + jcol) * 257 + 256];
  const float wf_n = Wih[(size_t)(512 + jcol) * 257 + 256];
  const float bg1j = bg1[jcol];
  const float bg2v = bg2[0];

  // ---- fill resident LDS weights (one time) ----
  for (int e = tid; e < HID * HID; e += NTHREADS) {
    const int r = e >> 8, c = e & 255;
    s_wg1[r * PITCH + c] = (__bf16)Wg1[e];
    s_whn[r * PITCH + c] = (__bf16)Whh[(size_t)(512 + r) * 256 + c];
  }
  for (int e = tid; e < HID; e += NTHREADS) s_w2[e] = Wg2[e];
  for (int e = tid; e < BT * PITCH; e += NTHREADS) s_h16[e] = (__bf16)0.0f;

  // per-lane owned state (rows mrow..mrow+7 of column jcol)
  float h8[8], acc8[8], hn8[8];
#pragma unroll
  for (int e = 0; e < 8; ++e) { h8[e] = 0.0f; acc8[e] = 0.0f; hn8[e] = 0.0f; }
  __syncthreads();

  // ================= sequence loop (persistent) =================
  for (int t = 0; t < SEQ; ++t) {
    const float* xt = x + ((size_t)t * BATCH + bbase) * INDIM;
    for (int e = tid; e < BT * INDIM; e += NTHREADS) {
      const int m = e >> 8, c = e & 255;
      s_x16[m * PITCH + c] = (__bf16)xt[(size_t)m * INDIM + c];
    }
    if (tid < BT) s_halt[tid] = 0.0f;
#pragma unroll
    for (int e = 0; e < 8; ++e) acc8[e] = 0.0f;
    __syncthreads();

    // ---- gi = x_t @ W_ih^T : ponder-invariant; W_ih must STREAM once per timestep.
    // 'tz' is an opaque per-iteration zero: blocks LICM from hoisting the 24 B-fragment
    // loads out of the t-loop (which would spill ~192 VGPRs to scratch).
    {
      int tz = 0;
      asm volatile("" : "+v"(tz));
      const __bf16* w0 = wih16 + (size_t)(      jcol) * 256 + tz;
      const __bf16* w1 = wih16 + (size_t)(256 + jcol) * 256 + tz;
      const __bf16* w2 = wih16 + (size_t)(512 + jcol) * 256 + tz;
      v8f gr = {}, gz = {}, gn = {};
#pragma unroll
      for (int k = 0; k < 8; ++k) {
        v16bf a = load_a_lds(s_x16, k, lane);
        gr = wmma_bf16(a, load_b_bf16(w0, k, lane), gr);
        gz = wmma_bf16(a, load_b_bf16(w1, k, lane), gz);
        gn = wmma_bf16(a, load_b_bf16(w2, k, lane), gn);
      }
#pragma unroll
      for (int e = 0; e < 8; ++e) {
        gstash[e]      = (_Float16)gr[e];
        gstash[8 + e]  = (_Float16)gz[e];
        gstash[16 + e] = (_Float16)gn[e];
      }
    }

    for (int pn = 0; pn < 5; ++pn) {
      const float flag = (pn > 0) ? 1.0f : 0.0f;

      // 'pz': opaque per-ponder-iteration zero; keeps the LDS-resident B-fragment loads
      // (s_whn, s_wg1) inside the pn loop instead of hoisted+spilled.
      int pz = 0;
      asm volatile("" : "+v"(pz));
      const __bf16* wnrow = s_whn + (size_t)jcol * PITCH + pz;
      const __bf16* wgrow = s_wg1 + (size_t)jcol * PITCH + pz;

      // ---- stage 1a: r/z h-part GEMM (register-resident B operands) ----
      v8f ar = {}, az = {};
#pragma unroll
      for (int k = 0; k < 8; ++k) {
        v16bf a = load_a_lds(s_h16, k, lane);
        ar = wmma_bf16(a, whr[k], ar);
        az = wmma_bf16(a, whz[k], az);
      }
      // ---- stage 1b: n-gate h-part GEMM (LDS-resident B operands) ----
      v8f ah = {};
#pragma unroll
      for (int k = 0; k < 8; ++k) {
        v16bf a = load_a_lds(s_h16, k, lane);
        ah = wmma_bf16(a, load_b_bf16(wnrow, k, lane), ah);
      }
      __syncthreads();  // all reads of s_h16 done before overwrite

      // ---- GRU elementwise: state in this lane's registers, gi from private stash ----
#pragma unroll
      for (int e = 0; e < 8; ++e) {
        const int m = mrow + e;
        const float hprev = h8[e];
        const float r  = sigmoidf_((float)gstash[e]      + ar[e] + b_r + flag * wf_r);
        const float z  = sigmoidf_((float)gstash[8 + e]  + az[e] + b_z + flag * wf_z);
        const float ng = tanhf((float)gstash[16 + e] + bi_n + flag * wf_n + r * (ah[e] + bh_n));
        const float hnew = (1.0f - z) * ng + z * hprev;
        const float hupd = (s_halt[m] < 0.99f) ? hnew : hprev;  // freeze halted rows
        hn8[e] = hnew;
        h8[e]  = hupd;
        s_h16[m * PITCH + jcol] = (__bf16)hupd;
      }
      __syncthreads();

      // ---- stage 2: g = gelu(h @ Wg1^T + bg1) ----
      v8f ag = {};
#pragma unroll
      for (int k = 0; k < 8; ++k) {
        v16bf a = load_a_lds(s_h16, k, lane);
        ag = wmma_bf16(a, load_b_bf16(wgrow, k, lane), ag);
      }
#pragma unroll
      for (int e = 0; e < 8; ++e) {
        const float v = ag[e] + bg1j;
        s_g16[(mrow + e) * HID + jcol] =
            (__bf16)(0.5f * v * (1.0f + erff(v * 0.70710678118654752f)));
      }
      __syncthreads();

      // ---- stage 3: halting probability per batch row ----
      if (tid < BT) {
        float sum = bg2v;
#pragma unroll 8
        for (int k = 0; k < HID; ++k) sum += (float)s_g16[tid * HID + k] * s_w2[k];
        const float p  = sigmoidf_(sum);
        const float hb = s_halt[tid];
        const float pr = (pn < 4) ? p : (1.0f - hb);
        const float pm = (hb < 0.99f) ? pr : 0.0f;
        s_pm[tid]   = pm;
        s_halt[tid] = hb + pm;
      }
      __syncthreads();

      // ---- accum += pm * h_new (register-resident accumulator) ----
#pragma unroll
      for (int e = 0; e < 8; ++e) acc8[e] += s_pm[mrow + e] * hn8[e];
    }

    // ---- emit outputs[t] for this batch tile ----
    float* ot = out + ((size_t)t * BATCH + bbase) * HID;
#pragma unroll
    for (int e = 0; e < 8; ++e) ot[(size_t)(mrow + e) * HID + jcol] = acc8[e];
  }
}

extern "C" void kernel_launch(void* const* d_in, const int* in_sizes, int n_in,
                              void* d_out, int out_size, void* d_ws, size_t ws_size,
                              hipStream_t stream) {
  (void)in_sizes; (void)n_in; (void)out_size; (void)ws_size;
  const float* x   = (const float*)d_in[0];
  const float* Wih = (const float*)d_in[1];
  const float* Whh = (const float*)d_in[2];
  const float* bih = (const float*)d_in[3];
  const float* bhh = (const float*)d_in[4];
  const float* Wg1 = (const float*)d_in[5];
  const float* bg1 = (const float*)d_in[6];
  const float* Wg2 = (const float*)d_in[7];
  const float* bg2 = (const float*)d_in[8];
  __bf16* wih16 = (__bf16*)d_ws;   // 768*256*2 = 393 KB of workspace
  float* out = (float*)d_out;

  hipLaunchKernelGGL(wih_to_bf16, dim3((768 * 256 + 255) / 256), dim3(256), 0, stream,
                     Wih, wih16);
  hipLaunchKernelGGL(act_gru_persistent, dim3(BATCH / BT), dim3(NTHREADS), 0, stream,
                     x, Wih, Whh, bih, bhh, Wg1, bg1, Wg2, bg2, wih16, out);
}
